// Qwen3_5Model_71038759075922
// MI455X (gfx1250) — compile-verified, matrix operand order checked by reading, not run on hardware
//
#include <hip/hip_runtime.h>
#include <hip/hip_bf16.h>
#include <math.h>

typedef __bf16 bf16;
typedef __attribute__((ext_vector_type(16))) __bf16 v16bf;
typedef __attribute__((ext_vector_type(8)))  __bf16 v8bf;
typedef __attribute__((ext_vector_type(8)))  float  v8f;

constexpr int kT = 2048;   // seq len
constexpr int kE = 1024;   // hidden
constexpr int kH = 16;     // q heads
constexpr int kD = 64;     // head dim
constexpr int kG = 4;      // kv heads
constexpr int kF = 4096;   // ffn
constexpr int kV = 32000;  // vocab
constexpr int kL = 4;      // layers
constexpr float kEPS = 1e-6f;
constexpr float kSCALE = 0.125f; // D^-0.5

// ---------------------------------------------------------------------------
// Fragment helpers (CDNA5 wave32 WMMA 16x16x32 bf16 layouts)
// A/B fragment, 16-bit: lane l -> row/col = l&15, khalf = l>>4,
// elems [0..7] = K = kh*8+0..7 ; elems [8..15] = K = 16+kh*8+0..7
// ---------------------------------------------------------------------------
static __device__ __forceinline__ v16bf load_frag(const bf16* row, int kh) {
  v8bf lo = *(const v8bf*)(row + kh * 8);
  v8bf hi = *(const v8bf*)(row + 16 + kh * 8);
  v16bf r;
#pragma unroll
  for (int i = 0; i < 8; ++i) { r[i] = lo[i]; r[8 + i] = hi[i]; }
  return r;
}

static __device__ __forceinline__ v8f wmma_bf16(v16bf a, v16bf b, v8f c) {
  return __builtin_amdgcn_wmma_f32_16x16x32_bf16(false, a, false, b, (short)0, c,
                                                 false, false);
}

// ---------------------------------------------------------------------------
// Embedding gather: x[t][e] = emb[tok[t]][e]
// ---------------------------------------------------------------------------
__global__ __launch_bounds__(256) void embed_kernel(const int* __restrict__ tok,
                                                    const float* __restrict__ emb,
                                                    float* __restrict__ x) {
  int t = blockIdx.x;
  const float* src = emb + (size_t)tok[t] * kE;
  float* dst = x + (size_t)t * kE;
  for (int i = threadIdx.x; i < kE; i += 256) dst[i] = src[i];
}

// ---------------------------------------------------------------------------
// RMSNorm over E=1024 -> bf16 out. out = x * rsqrt(mean(x^2)+eps) * (1+w)
// ---------------------------------------------------------------------------
__global__ __launch_bounds__(256) void rmsnorm_kernel(const float* __restrict__ x,
                                                      const float* __restrict__ w,
                                                      bf16* __restrict__ out) {
  int t = blockIdx.x, tid = threadIdx.x;
  const float* row = x + (size_t)t * kE;
  float v[4], ss = 0.f;
#pragma unroll
  for (int i = 0; i < 4; ++i) { v[i] = row[tid + i * 256]; ss += v[i] * v[i]; }
#pragma unroll
  for (int m = 16; m >= 1; m >>= 1) ss += __shfl_xor(ss, m, 32);
  __shared__ float red[8];
  __shared__ float srs;
  if ((tid & 31) == 0) red[tid >> 5] = ss;
  __syncthreads();
  if (tid == 0) {
    float tot = 0.f;
#pragma unroll
    for (int i = 0; i < 8; ++i) tot += red[i];
    srs = rsqrtf(tot / (float)kE + kEPS);
  }
  __syncthreads();
  float rs = srs;
#pragma unroll
  for (int i = 0; i < 4; ++i) {
    int e = tid + i * 256;
    out[(size_t)t * kE + e] = (bf16)(v[i] * rs * (1.f + w[e]));
  }
}

// ---------------------------------------------------------------------------
// Generic GEMM: C[M][N] (f32) = A_bf16[M][K] @ B_f32[K][N] (+ resid)
// Block tile 128x64, 8 waves each computing 32x32 (4 WMMAs / 32-k step).
// B tile staged transposed in LDS (f32 -> bf16) with one ds_store_b128 per
// thread per tile; next tile prefetched via global_prefetch_b8.
// K is a compile-time template parameter (lda == K for all call sites).
// ---------------------------------------------------------------------------
template <int K>
__global__ __launch_bounds__(256) void gemm_kernel(const bf16* __restrict__ A,
                                                   const float* __restrict__ B, int ldb,
                                                   float* __restrict__ C,
                                                   const float* __restrict__ resid,
                                                   int N) {
  __shared__ __attribute__((aligned(16))) bf16 bt[64 * 32]; // [n_local][k_local]
  int tid = threadIdx.x;
  int wv = tid >> 5, lane = tid & 31;
  int nl = lane & 15, half = lane >> 4;
  int m0 = blockIdx.y * 128, n0 = blockIdx.x * 64;
  int mw = m0 + (wv & 3) * 32;           // wave row base (32 rows)
  int nwl = (wv >> 2) * 32;              // wave col offset within 64-tile
  const int nn = tid & 63;               // staging column
  const int kb = (tid >> 6) * 8;         // staging k-slice base

  v8f c00 = {}, c01 = {}, c10 = {}, c11 = {};

  const bf16* arow0 = A + (size_t)(mw + nl) * K;
  const bf16* arow1 = A + (size_t)(mw + 16 + nl) * K;

  for (int k0 = 0; k0 < K; k0 += 32) {
    __syncthreads();
    {
      const float* bp = B + (size_t)(k0 + kb) * ldb + n0 + nn;
      float f[8];
#pragma unroll
      for (int j = 0; j < 8; ++j) f[j] = bp[(size_t)j * ldb];
      v8bf pk;
#pragma unroll
      for (int j = 0; j < 8; ++j) pk[j] = (bf16)f[j];
      *(v8bf*)(bt + nn * 32 + kb) = pk;
      if (k0 + 32 < K) __builtin_prefetch(bp + (size_t)32 * ldb, 0, 3);
    }
    __syncthreads();

    v16bf a0 = load_frag(arow0 + k0, half);
    v16bf a1 = load_frag(arow1 + k0, half);
    v16bf b0 = load_frag(bt + (nwl + nl) * 32, half);
    v16bf b1 = load_frag(bt + (nwl + 16 + nl) * 32, half);
    c00 = wmma_bf16(a0, b0, c00);
    c01 = wmma_bf16(a0, b1, c01);
    c10 = wmma_bf16(a1, b0, c10);
    c11 = wmma_bf16(a1, b1, c11);
  }

#pragma unroll
  for (int r = 0; r < 8; ++r) {
    int row0 = mw + half * 8 + r;
    int row1 = row0 + 16;
    size_t o00 = (size_t)row0 * N + n0 + nwl + nl;
    size_t o01 = o00 + 16;
    size_t o10 = (size_t)row1 * N + n0 + nwl + nl;
    size_t o11 = o10 + 16;
    float v00 = c00[r], v01 = c01[r], v10 = c10[r], v11 = c11[r];
    if (resid) {
      v00 += resid[o00]; v01 += resid[o01];
      v10 += resid[o10]; v11 += resid[o11];
    }
    C[o00] = v00; C[o01] = v01; C[o10] = v10; C[o11] = v11;
  }
}

// ---------------------------------------------------------------------------
// QK-norm + partial RoPE (ROT=16) + layout prep.
// qg   f32 [T][H*128]  (per head: 64 q | 64 gate)  -> qh bf16 [H][T][64]
// kbuf f32 [T][G*64]                               -> khb bf16 [G][T][64]
// vbuf f32 [T][G*64]                               -> vT  bf16 [G][64][T]
// ---------------------------------------------------------------------------
__global__ __launch_bounds__(256) void qkprep_kernel(const float* __restrict__ qg,
                                                     const float* __restrict__ kbuf,
                                                     const float* __restrict__ vbuf,
                                                     const float* __restrict__ qn,
                                                     const float* __restrict__ kn,
                                                     bf16* __restrict__ qh,
                                                     bf16* __restrict__ khb,
                                                     bf16* __restrict__ vT) {
  int t = blockIdx.x;
  int wv = threadIdx.x >> 5, lane = threadIdx.x & 31;
  float ang = (float)t * __powf(10000.f, -(float)(lane & 7) / 8.f);
  float cs = __cosf(ang), sn = __sinf(ang);

  // ---- Q: waves 0..7 handle heads wv and wv+8 ----
#pragma unroll
  for (int rep = 0; rep < 2; ++rep) {
    int h = wv + rep * 8;
    const float* src = qg + (size_t)t * (2 * kH * kD) + h * 128;
    float x0 = src[lane], x1 = src[lane + 32];
    float ss = x0 * x0 + x1 * x1;
#pragma unroll
    for (int m = 16; m >= 1; m >>= 1) ss += __shfl_xor(ss, m, 32);
    float rs = rsqrtf(ss / (float)kD + kEPS);
    float y0 = x0 * rs * (1.f + qn[lane]);
    float y1 = x1 * rs * (1.f + qn[lane + 32]);
    float part = __shfl_xor(y0, 8, 32);
    if (lane < 16) {
      float rot = (lane < 8) ? -part : part;
      y0 = y0 * cs + rot * sn;
    }
    bf16* dst = qh + ((size_t)h * kT + t) * kD;
    dst[lane] = (bf16)y0;
    dst[lane + 32] = (bf16)y1;
  }

  if (wv < 4) {
    // ---- K: waves 0..3 handle group wv ----
    int g = wv;
    const float* src = kbuf + (size_t)t * (kG * kD) + g * kD;
    float x0 = src[lane], x1 = src[lane + 32];
    float ss = x0 * x0 + x1 * x1;
#pragma unroll
    for (int m = 16; m >= 1; m >>= 1) ss += __shfl_xor(ss, m, 32);
    float rs = rsqrtf(ss / (float)kD + kEPS);
    float y0 = x0 * rs * (1.f + kn[lane]);
    float y1 = x1 * rs * (1.f + kn[lane + 32]);
    float part = __shfl_xor(y0, 8, 32);
    if (lane < 16) {
      float rot = (lane < 8) ? -part : part;
      y0 = y0 * cs + rot * sn;
    }
    bf16* dst = khb + ((size_t)g * kT + t) * kD;
    dst[lane] = (bf16)y0;
    dst[lane + 32] = (bf16)y1;
  } else {
    // ---- V transpose: waves 4..7 handle group wv-4 ----
    int g = wv - 4;
    const float* src = vbuf + (size_t)t * (kG * kD) + g * kD;
    vT[((size_t)g * kD + lane) * kT + t] = (bf16)src[lane];
    vT[((size_t)g * kD + lane + 32) * kT + t] = (bf16)src[lane + 32];
  }
}

// ---------------------------------------------------------------------------
// Flash attention (causal, GQA) with WMMA.
// Computes S^T = K @ Q^T so the softmaxed P^T repacks in-lane into the
// B-fragment for O^T = V^T @ P^T. One wave per 16-query tile per head.
// Epilogue fuses /l, sigmoid gate, bf16 store into attnA[T][H*D].
// grid = (T/64, H), block = 128 (4 waves).
// ---------------------------------------------------------------------------
__global__ __launch_bounds__(128) void attn_kernel(const bf16* __restrict__ qh,
                                                   const bf16* __restrict__ khb,
                                                   const bf16* __restrict__ vT,
                                                   const float* __restrict__ qg,
                                                   bf16* __restrict__ attnA) {
  int wv = threadIdx.x >> 5, lane = threadIdx.x & 31;
  int nl = lane & 15, half = lane >> 4;
  int head = blockIdx.y;
  int g = head >> 2;                    // H/G = 4
  int q0 = (blockIdx.x * 4 + wv) * 16;
  int q = q0 + nl;

  const bf16* qbase = qh + ((size_t)head * kT + q0 + nl) * kD;
  v16bf bq0 = load_frag(qbase, half);        // feats 0..31
  v16bf bq1 = load_frag(qbase + 32, half);   // feats 32..63

  v8f o[4] = {};
  float mrun = -1e30f, lrun = 0.f;

  int ntiles = (q0 + 16 + 31) / 32;
  for (int kt = 0; kt < ntiles; ++kt) {
    int kb = kt * 32;
    // S^T tile: 32 keys x 16 queries = two C fragments
    v8f s0 = {}, s1 = {};
    const bf16* k0p = khb + ((size_t)g * kT + kb + nl) * kD;
    const bf16* k1p = khb + ((size_t)g * kT + kb + 16 + nl) * kD;
    s0 = wmma_bf16(load_frag(k0p, half), bq0, s0);
    s0 = wmma_bf16(load_frag(k0p + 32, half), bq1, s0);
    s1 = wmma_bf16(load_frag(k1p, half), bq0, s1);
    s1 = wmma_bf16(load_frag(k1p + 32, half), bq1, s1);

    // scale + causal mask; each lane holds 16 keys for its query column
    float p[16];
    float tmax = -1e30f;
#pragma unroll
    for (int j = 0; j < 8; ++j) {
      int key0 = kb + half * 8 + j;
      int key1 = kb + 16 + half * 8 + j;
      float x0 = (key0 <= q) ? s0[j] * kSCALE : -1e30f;
      float x1 = (key1 <= q) ? s1[j] * kSCALE : -1e30f;
      p[j] = x0; p[8 + j] = x1;
      tmax = fmaxf(tmax, fmaxf(x0, x1));
    }
    tmax = fmaxf(tmax, __shfl_xor(tmax, 16, 32));
    float mnew = fmaxf(mrun, tmax);
    float alpha = __expf(mrun - mnew);
    float psum = 0.f;
    v16bf bp;                            // P^T B-fragment, pure in-lane repack
#pragma unroll
    for (int j = 0; j < 16; ++j) {
      float e = __expf(p[j] - mnew);
      psum += e;
      bp[j] = (bf16)e;
    }
    lrun = lrun * alpha + psum + __shfl_xor(psum, 16, 32);
    mrun = mnew;
#pragma unroll
    for (int f = 0; f < 4; ++f)
#pragma unroll
      for (int r = 0; r < 8; ++r) o[f][r] *= alpha;

    const bf16* vb = vT + ((size_t)g * kD + nl) * kT + kb;
#pragma unroll
    for (int f = 0; f < 4; ++f)
      o[f] = wmma_bf16(load_frag(vb + (size_t)f * 16 * kT, half), bp, o[f]);
  }

  // epilogue: O^T lane holds query q, rows d = f*16 + half*8 + r
  float inv = 1.f / lrun;
  const float* gbase = qg + (size_t)q * (2 * kH * kD) + head * 128 + 64;
  bf16* obase = attnA + (size_t)q * (kH * kD) + head * kD;
#pragma unroll
  for (int f = 0; f < 4; ++f)
#pragma unroll
    for (int r = 0; r < 8; ++r) {
      int d = f * 16 + half * 8 + r;
      float val = o[f][r] * inv;
      float gt = gbase[d];
      val = val / (1.f + __expf(-gt));   // * sigmoid(gate)
      obase[d] = (bf16)val;
    }
}

// ---------------------------------------------------------------------------
// SwiGLU: u = silu(a) * b -> bf16
// ---------------------------------------------------------------------------
__global__ __launch_bounds__(256) void swiglu_kernel(const float* __restrict__ a,
                                                     const float* __restrict__ b,
                                                     bf16* __restrict__ u, int n) {
  int i = blockIdx.x * 256 + threadIdx.x;
  if (i < n) {
    float av = a[i];
    u[i] = (bf16)((av / (1.f + __expf(-av))) * b[i]);
  }
}

// ---------------------------------------------------------------------------
// Host orchestration
// ---------------------------------------------------------------------------
extern "C" void kernel_launch(void* const* d_in, const int* in_sizes, int n_in,
                              void* d_out, int out_size, void* d_ws, size_t ws_size,
                              hipStream_t stream) {
  (void)in_sizes; (void)n_in; (void)out_size; (void)ws_size;
  const int*   tok  = (const int*)d_in[0];
  const float* emb  = (const float*)d_in[1];
  const float* Wq   = (const float*)d_in[2];
  const float* Wk   = (const float*)d_in[3];
  const float* Wv   = (const float*)d_in[4];
  const float* Wo   = (const float*)d_in[5];
  const float* qn   = (const float*)d_in[6];
  const float* kn   = (const float*)d_in[7];
  const float* n1   = (const float*)d_in[8];
  const float* n2   = (const float*)d_in[9];
  const float* f1   = (const float*)d_in[10];
  const float* f2   = (const float*)d_in[11];
  const float* f3   = (const float*)d_in[12];
  const float* fn   = (const float*)d_in[13];
  const float* head = (const float*)d_in[14];
  float* out = (float*)d_out;

  char* base = (char*)d_ws;
  size_t off = 0;
  auto take = [&](size_t bytes) -> void* {
    void* p = base + off;
    off = (off + bytes + 255) & ~(size_t)255;
    return p;
  };
  float* x     = (float*)take((size_t)kT * kE * 4);
  bf16*  h     = (bf16*) take((size_t)kT * kE * 2);
  float* qg    = (float*)take((size_t)kT * 2 * kH * kD * 4);
  float* kbuf  = (float*)take((size_t)kT * kG * kD * 4);
  float* vbuf  = (float*)take((size_t)kT * kG * kD * 4);
  bf16*  qhb   = (bf16*) take((size_t)kH * kT * kD * 2);
  bf16*  khb   = (bf16*) take((size_t)kG * kT * kD * 2);
  bf16*  vTb   = (bf16*) take((size_t)kG * kD * kT * 2);
  bf16*  attnA = (bf16*) take((size_t)kT * kH * kD * 2);
  float* ffa   = (float*)take((size_t)kT * kF * 4);
  float* ffb   = (float*)take((size_t)kT * kF * 4);
  bf16*  u     = (bf16*) take((size_t)kT * kF * 2);

  const dim3 blk(256);
  const int mgrid = kT / 128;            // 16

  embed_kernel<<<kT, 256, 0, stream>>>(tok, emb, x);

  for (int l = 0; l < kL; ++l) {
    const float* wq = Wq + (size_t)l * kE * 2 * kH * kD;
    const float* wk = Wk + (size_t)l * kE * kG * kD;
    const float* wv = Wv + (size_t)l * kE * kG * kD;
    const float* wo = Wo + (size_t)l * kH * kD * kE;
    const float* g1 = f1 + (size_t)l * kE * kF;
    const float* g2 = f2 + (size_t)l * kE * kF;
    const float* g3 = f3 + (size_t)l * kF * kE;

    rmsnorm_kernel<<<kT, 256, 0, stream>>>(x, n1 + (size_t)l * kE, h);

    gemm_kernel<kE><<<dim3(2 * kH * kD / 64, mgrid), blk, 0, stream>>>(
        h, wq, 2 * kH * kD, qg, nullptr, 2 * kH * kD);
    gemm_kernel<kE><<<dim3(kG * kD / 64, mgrid), blk, 0, stream>>>(
        h, wk, kG * kD, kbuf, nullptr, kG * kD);
    gemm_kernel<kE><<<dim3(kG * kD / 64, mgrid), blk, 0, stream>>>(
        h, wv, kG * kD, vbuf, nullptr, kG * kD);

    qkprep_kernel<<<kT, 256, 0, stream>>>(qg, kbuf, vbuf,
                                          qn + (size_t)l * kD, kn + (size_t)l * kD,
                                          qhb, khb, vTb);

    attn_kernel<<<dim3(kT / 64, kH), 128, 0, stream>>>(qhb, khb, vTb, qg, attnA);

    gemm_kernel<kE><<<dim3(kE / 64, mgrid), blk, 0, stream>>>(
        attnA, wo, kE, x, x, kE);

    rmsnorm_kernel<<<kT, 256, 0, stream>>>(x, n2 + (size_t)l * kE, h);

    gemm_kernel<kE><<<dim3(kF / 64, mgrid), blk, 0, stream>>>(
        h, g1, kF, ffa, nullptr, kF);
    gemm_kernel<kE><<<dim3(kF / 64, mgrid), blk, 0, stream>>>(
        h, g2, kF, ffb, nullptr, kF);

    swiglu_kernel<<<(kT * kF) / 256, 256, 0, stream>>>(ffa, ffb, u, kT * kF);

    gemm_kernel<kF><<<dim3(kE / 64, mgrid), blk, 0, stream>>>(
        u, g3, kE, x, x, kE);
  }

  rmsnorm_kernel<<<kT, 256, 0, stream>>>(x, fn, h);
  gemm_kernel<kE><<<dim3(kV / 64, mgrid), blk, 0, stream>>>(
      h, head, kV, out, nullptr, kV);
}
